// JetBlock_89627377533673
// MI455X (gfx1250) — matrix operand order repaired and checked by reading, not compiled, not run on hardware
//
#include <hip/hip_runtime.h>
#include <hip/hip_bf16.h>

// ---------------------------------------------------------------------------
// JetBlock for MI455X (gfx1250, wave32, WMMA).
// Dense GEMMs (~380 GFLOP) -> bf16 V_WMMA_F32_16X16X32_BF16, fp32 accum.
// Weights are transposed at bf16-convert time so BOTH wmma operands are
// K-contiguous per lane: all LDS fragment traffic is ds_load_b128 (no scalar
// u16 gather), LDS rows padded 32->40 bf16 to kill 4-way bank conflicts, and
// the K-loop is double-buffered so global staging overlaps the matrix pipe.
// Sequential gated-delta recurrence keeps its 128x128 fp32 state in LDS.
// ---------------------------------------------------------------------------

typedef __bf16 bf16_t;
typedef __attribute__((ext_vector_type(16))) __bf16 v16bf;
typedef __attribute__((ext_vector_type(8)))  __bf16 v8bf;
typedef __attribute__((ext_vector_type(8)))  float  v8f;

#define BB   2
#define TT   2048
#define HID  2048
#define HH   16
#define WW   4
#define MROWS (BB*TT)          // 4096
#define LPAD 40                // LDS tile row stride in bf16 (32 data + 8 pad)

__device__ __forceinline__ float silu_f(float x) { return x / (1.0f + __expf(-x)); }
__device__ __forceinline__ float sigmoid_f(float x) { return 1.0f / (1.0f + __expf(-x)); }

// ---------------------------------------------------------------------------
// bf16 WMMA GEMM: C[M,N] = A[M,K] * Bt[N,K]^T (+bias[N]).
// A row-major [M,K] bf16; Bt is the TRANSPOSED weight [N,K] bf16; C fp32.
// Block tile 128x128, BK=32, 256 threads = 8 waves (2x4); per wave 64x32
// (4x2 wmma accumulators). Double-buffered LDS staging.
// ---------------------------------------------------------------------------
__global__ __launch_bounds__(256) void gemm_bf16_wmma(
    const bf16_t* __restrict__ A, const bf16_t* __restrict__ Bt,
    float* __restrict__ C, const float* __restrict__ bias,
    int M, int N, int K)
{
    __shared__ bf16_t sA[2][128][LPAD];
    __shared__ bf16_t sB[2][128][LPAD];

    const int tid  = threadIdx.x;
    const int lane = tid & 31;
    const int wave = tid >> 5;
    const int waveRow = (wave >> 2) * 64;   // 0 or 64
    const int waveCol = (wave & 3) * 32;    // 0,32,64,96
    const int r    = lane & 15;
    const int half = lane >> 4;

    const int blockRow = blockIdx.y * 128;
    const int blockCol = blockIdx.x * 128;

    // staging: 512 16-byte chunks per tile, 2 per thread. chunk c: row=c>>2,
    // col=(c&3)*8. Same pattern for A (rows of A) and B (rows of Bt).
    const int c0  = tid,        c1  = tid + 256;
    const int sr0 = c0 >> 2,    sc0 = (c0 & 3) * 8;
    const int sr1 = c1 >> 2,    sc1 = (c1 & 3) * 8;

    const bf16_t* Ap = A  + (size_t)blockRow * K;
    const bf16_t* Bp = Bt + (size_t)blockCol * K;

    v8f acc[4][2] = {};

    // prologue: stage tile 0
    v8bf ra0 = *(const v8bf*)&Ap[(size_t)sr0 * K + sc0];
    v8bf ra1 = *(const v8bf*)&Ap[(size_t)sr1 * K + sc1];
    v8bf rb0 = *(const v8bf*)&Bp[(size_t)sr0 * K + sc0];
    v8bf rb1 = *(const v8bf*)&Bp[(size_t)sr1 * K + sc1];
    *(v8bf*)&sA[0][sr0][sc0] = ra0;  *(v8bf*)&sA[0][sr1][sc1] = ra1;
    *(v8bf*)&sB[0][sr0][sc0] = rb0;  *(v8bf*)&sB[0][sr1][sc1] = rb1;

    int cur = 0;
    for (int k0 = 0; k0 < K; k0 += 32) {
        __syncthreads();   // buffer `cur` fully staged

        const bool more = (k0 + 32) < K;
        if (more) {        // issue next tile's global loads (overlap with wmma)
            ra0 = *(const v8bf*)&Ap[(size_t)sr0 * K + (k0 + 32) + sc0];
            ra1 = *(const v8bf*)&Ap[(size_t)sr1 * K + (k0 + 32) + sc1];
            rb0 = *(const v8bf*)&Bp[(size_t)sr0 * K + (k0 + 32) + sc0];
            rb1 = *(const v8bf*)&Bp[(size_t)sr1 * K + (k0 + 32) + sc1];
        }

        // A frags: lane<16 -> row M=r, K 0..7 & 16..23; lane>=16 -> K 8..15 & 24..31
        v16bf afrag[4];
        #pragma unroll
        for (int m0 = 0; m0 < 4; ++m0) {
            int row = waveRow + m0 * 16 + r;
            v8bf lo = *(const v8bf*)&sA[cur][row][half * 8];
            v8bf hi = *(const v8bf*)&sA[cur][row][16 + half * 8];
            #pragma unroll
            for (int i = 0; i < 8; ++i) { afrag[m0][i] = lo[i]; afrag[m0][8 + i] = hi[i]; }
        }
        // B frags: lane r holds K 0..7 & 16..23 of column r, lane r+16 holds K 8..15 & 24..31;
        // K-contiguous in sB rows -> two ds_load_b128 per fragment.
        v16bf bfrag[2];
        #pragma unroll
        for (int n0 = 0; n0 < 2; ++n0) {
            int col = waveCol + n0 * 16 + r;
            v8bf lo = *(const v8bf*)&sB[cur][col][half * 8];
            v8bf hi = *(const v8bf*)&sB[cur][col][16 + half * 8];
            #pragma unroll
            for (int i = 0; i < 8; ++i) { bfrag[n0][i] = lo[i]; bfrag[n0][8 + i] = hi[i]; }
        }

        #pragma unroll
        for (int m0 = 0; m0 < 4; ++m0)
            #pragma unroll
            for (int n0 = 0; n0 < 2; ++n0)
                acc[m0][n0] = __builtin_amdgcn_wmma_f32_16x16x32_bf16(
                    false, afrag[m0], false, bfrag[n0],
                    (short)0, acc[m0][n0], false, false);

        if (more) {        // stage next tile into the other buffer
            int nxt = cur ^ 1;
            *(v8bf*)&sA[nxt][sr0][sc0] = ra0;  *(v8bf*)&sA[nxt][sr1][sc1] = ra1;
            *(v8bf*)&sB[nxt][sr0][sc0] = rb0;  *(v8bf*)&sB[nxt][sr1][sc1] = rb1;
        }
        cur ^= 1;
    }

    // C/D layout: VGPR j -> M = j + 8*half, N = lane%16 (per 16x16 tile)
    #pragma unroll
    for (int m0 = 0; m0 < 4; ++m0)
        #pragma unroll
        for (int n0 = 0; n0 < 2; ++n0)
            #pragma unroll
            for (int j = 0; j < 8; ++j) {
                int row = blockRow + waveRow + m0 * 16 + half * 8 + j;
                int col = blockCol + waveCol + n0 * 16 + r;
                float v = acc[m0][n0][j];
                if (bias) v += bias[col];
                C[(size_t)row * N + col] = v;
            }
}

// ---------------------------------------------------------------------------
// fp32 [K,N] -> bf16 TRANSPOSED [N,K] (tiled, coalesced both sides)
// ---------------------------------------------------------------------------
__global__ __launch_bounds__(256) void k_cvt_bf16_T(
    const float* __restrict__ s, bf16_t* __restrict__ d, int K, int N)
{
    __shared__ float tile[32][33];
    int nx = blockIdx.x * 32 + threadIdx.x;            // N index (read)
    #pragma unroll
    for (int j = 0; j < 4; ++j) {
        int ky = blockIdx.y * 32 + threadIdx.y + j * 8;
        tile[threadIdx.y + j * 8][threadIdx.x] = s[(size_t)ky * N + nx];
    }
    __syncthreads();
    int kx = blockIdx.y * 32 + threadIdx.x;            // K index (write)
    #pragma unroll
    for (int j = 0; j < 4; ++j) {
        int ny = blockIdx.x * 32 + threadIdx.y + j * 8;
        d[(size_t)ny * K + kx] = (bf16_t)tile[threadIdx.x][threadIdx.y + j * 8];
    }
}

// ---------------------------------------------------------------------------
// elementwise helpers
// ---------------------------------------------------------------------------
__global__ void k_cvt_bf16(const float* __restrict__ s, bf16_t* __restrict__ d, int n) {
    int i = blockIdx.x * blockDim.x + threadIdx.x;
    if (i < n) d[i] = (bf16_t)s[i];
}
__global__ void k_silu_cvt_bf16(const float* __restrict__ s, bf16_t* __restrict__ d, int n) {
    int i = blockIdx.x * blockDim.x + threadIdx.x;
    if (i < n) d[i] = (bf16_t)silu_f(s[i]);
}
// gi[row, 0:2048] = q, gi[row, 2048:4096] = k  (bf16)
__global__ void k_concat_qk(const float* __restrict__ q, const float* __restrict__ k,
                            bf16_t* __restrict__ gi) {
    int i = blockIdx.x * blockDim.x + threadIdx.x;   // MROWS*HID
    if (i >= MROWS * HID) return;
    int row = i / HID, c = i % HID;
    gi[(size_t)row * (2 * HID) + c]       = (bf16_t)q[i];
    gi[(size_t)row * (2 * HID) + HID + c] = (bf16_t)k[i];
}

// dynamic causal short conv + SiLU
__global__ void k_conv_silu(const float* __restrict__ kern, const float* __restrict__ v,
                            float* __restrict__ out) {
    size_t idx = (size_t)blockIdx.x * blockDim.x + threadIdx.x; // B*T*2048
    if (idx >= (size_t)MROWS * 2048) return;
    int d  = (int)(idx & 2047);
    int bt = (int)(idx >> 11);
    int t = bt % TT, b = bt / TT;
    const float4 kw = *(const float4*)&kern[idx * 4];
    float kv[4] = {kw.x, kw.y, kw.z, kw.w};
    float acc = 0.f;
    #pragma unroll
    for (int w = 0; w < WW; ++w) {
        int ts = t - (WW - 1) + w;
        if (ts >= 0) acc += kv[w] * v[((size_t)(b * TT + ts) << 11) + d];
    }
    out[idx] = silu_f(acc);
}

// beta = sigmoid(x@Wb), decay = exp(-exp(A_log)*softplus(x@Wa + dt_bias))  (N=16)
__global__ void k_gating(const float* __restrict__ x, const float* __restrict__ Wb,
                         const float* __restrict__ Wa, const float* __restrict__ dtb,
                         const float* __restrict__ Alog,
                         float* __restrict__ beta, float* __restrict__ decay) {
    int idx = blockIdx.x * blockDim.x + threadIdx.x;   // MROWS*H
    if (idx >= MROWS * HH) return;
    int h = idx & (HH - 1), row = idx / HH;
    const float* xr = &x[(size_t)row * HID];
    float db = 0.f, da = 0.f;
    for (int kk = 0; kk < HID; ++kk) {
        float xv = xr[kk];
        db += xv * Wb[kk * HH + h];
        da += xv * Wa[kk * HH + h];
    }
    beta[idx] = sigmoid_f(db);
    float a = da + dtb[h];
    float sp = (a > 20.f) ? a : __logf(1.f + __expf(a));
    decay[idx] = __expf(-__expf(Alog[h]) * sp);
}

// in-place L2 normalize rows of 128 (one wave32 per row, 8 rows/block)
__global__ __launch_bounds__(256) void k_l2norm(float* __restrict__ p, int nrows) {
    int wave = threadIdx.x >> 5, lane = threadIdx.x & 31;
    int row = blockIdx.x * 8 + wave;
    if (row >= nrows) return;
    float* q = &p[(size_t)row * 128];
    float v0 = q[lane], v1 = q[lane + 32], v2 = q[lane + 64], v3 = q[lane + 96];
    float ss = v0 * v0 + v1 * v1 + v2 * v2 + v3 * v3;
    #pragma unroll
    for (int off = 16; off; off >>= 1) ss += __shfl_xor(ss, off, 32);
    float inv = 1.0f / fmaxf(sqrtf(ss), 1e-12f);
    q[lane] = v0 * inv; q[lane + 32] = v1 * inv; q[lane + 64] = v2 * inv; q[lane + 96] = v3 * inv;
}

// gated delta rule: one block per (b,h), state 128x128 fp32 in LDS (stride 129)
__global__ __launch_bounds__(128) void k_recurrence(
    const float* __restrict__ q, const float* __restrict__ k,
    const float* __restrict__ v, const float* __restrict__ decay,
    const float* __restrict__ beta, float* __restrict__ o) {
    __shared__ float S[128 * 129];
    __shared__ float qs[128], ks[128];
    int b = blockIdx.x >> 4, h = blockIdx.x & 15;
    int dv = threadIdx.x;
    for (int i = dv; i < 128 * 129; i += 128) S[i] = 0.f;
    __syncthreads();
    for (int t = 0; t < TT; ++t) {
        size_t row  = (size_t)(b * TT + t);
        size_t base = (row << 11) + h * 128;
        qs[dv] = q[base + dv];
        ks[dv] = k[base + dv];
        __syncthreads();
        float vv  = v[base + dv];
        float dec = decay[row * HH + h];
        float bet = beta[row * HH + h];
        float ot = 0.f, sk = 0.f;
        #pragma unroll 8
        for (int dk = 0; dk < 128; ++dk) {
            float s = S[dk * 129 + dv];
            ot += s * qs[dk];
            sk += s * ks[dk];
        }
        float delta = (vv - sk) * bet;           // fold beta into delta
        #pragma unroll 8
        for (int dk = 0; dk < 128; ++dk)
            S[dk * 129 + dv] = dec * S[dk * 129 + dv] + ks[dk] * delta;
        o[base + dv] = ot;
        __syncthreads();
    }
}

// per-head RMSNorm * norm_weight * silu(gate) -> bf16
__global__ __launch_bounds__(256) void k_rmsnorm_gate(
    const float* __restrict__ o, const float* __restrict__ gate,
    const float* __restrict__ nw, bf16_t* __restrict__ y, int nrows) {
    int wave = threadIdx.x >> 5, lane = threadIdx.x & 31;
    int row = blockIdx.x * 8 + wave;
    if (row >= nrows) return;
    const float* op = &o[(size_t)row * 128];
    float v0 = op[lane], v1 = op[lane + 32], v2 = op[lane + 64], v3 = op[lane + 96];
    float ss = v0 * v0 + v1 * v1 + v2 * v2 + v3 * v3;
    #pragma unroll
    for (int off = 16; off; off >>= 1) ss += __shfl_xor(ss, off, 32);
    float r = rsqrtf(ss * (1.0f / 128.0f) + 1e-6f);
    float vals[4] = {v0, v1, v2, v3};
    #pragma unroll
    for (int i = 0; i < 4; ++i) {
        int d = lane + 32 * i;
        float g = gate[(size_t)row * 128 + d];
        y[(size_t)row * 128 + d] = (bf16_t)(vals[i] * r * nw[d] * silu_f(g));
    }
}

// ---------------------------------------------------------------------------
// host orchestration
// ---------------------------------------------------------------------------
extern "C" void kernel_launch(void* const* d_in, const int* in_sizes, int n_in,
                              void* d_out, int out_size, void* d_ws, size_t ws_size,
                              hipStream_t stream) {
    (void)in_sizes; (void)n_in; (void)out_size; (void)ws_size;
    const float* x     = (const float*)d_in[0];
    const float* Wq    = (const float*)d_in[1];
    const float* Wk    = (const float*)d_in[2];
    const float* Wv    = (const float*)d_in[3];
    const float* Wb    = (const float*)d_in[4];
    const float* Wa    = (const float*)d_in[5];
    const float* dtb   = (const float*)d_in[6];
    const float* Alog  = (const float*)d_in[7];
    const float* gw1   = (const float*)d_in[8];
    const float* gw2   = (const float*)d_in[9];
    const float* gb2   = (const float*)d_in[10];
    const float* nw    = (const float*)d_in[11];
    const float* Wg    = (const float*)d_in[12];
    const float* Wo    = (const float*)d_in[13];
    float* out = (float*)d_out;

    // bump allocator over d_ws (total ~550 MB)
    char* base = (char*)d_ws;
    size_t off = 0;
    auto alloc = [&](size_t bytes) -> void* {
        void* p = base + off;
        off += (bytes + 255) & ~(size_t)255;
        return p;
    };
    const size_t MH = (size_t)MROWS * HID;          // 4096*2048
    bf16_t* xb   = (bf16_t*)alloc(MH * 2);
    bf16_t* Wqt  = (bf16_t*)alloc((size_t)HID * HID * 2);
    bf16_t* Wkt  = (bf16_t*)alloc((size_t)HID * HID * 2);
    bf16_t* Wvt  = (bf16_t*)alloc((size_t)HID * HID * 2);
    bf16_t* Wgt  = (bf16_t*)alloc((size_t)HID * HID * 2);
    bf16_t* Wot  = (bf16_t*)alloc((size_t)HID * HID * 2);
    bf16_t* g1t  = (bf16_t*)alloc((size_t)(2 * HID) * HID * 2);
    bf16_t* g2t  = (bf16_t*)alloc((size_t)HID * (HID * WW) * 2);
    float*  qf   = (float*)alloc(MH * 4);
    float*  kf   = (float*)alloc(MH * 4);
    float*  vf   = (float*)alloc(MH * 4);
    float*  gatef= (float*)alloc(MH * 4);
    bf16_t* gi   = (bf16_t*)alloc((size_t)MROWS * 2 * HID * 2);
    float*  h1f  = (float*)alloc(MH * 4);
    bf16_t* h1b  = (bf16_t*)alloc(MH * 2);
    float*  kernf= (float*)alloc((size_t)MROWS * HID * WW * 4);   // 128 MB
    float*  vs   = (float*)alloc(MH * 4);
    float*  betaf= (float*)alloc((size_t)MROWS * HH * 4);
    float*  decf = (float*)alloc((size_t)MROWS * HH * 4);
    float*  of   = (float*)alloc(MH * 4);
    bf16_t* yb   = (bf16_t*)alloc(MH * 2);

    auto cvtT = [&](const float* s, bf16_t* d, int K, int N) {  // [K,N] -> bf16 [N,K]
        dim3 g(N / 32, K / 32), b(32, 8);
        k_cvt_bf16_T<<<g, b, 0, stream>>>(s, d, K, N);
    };
    auto gemm = [&](const bf16_t* A, const bf16_t* Bt, float* C, const float* bias,
                    int M, int N, int K) {
        dim3 g(N / 128, M / 128);
        gemm_bf16_wmma<<<g, 256, 0, stream>>>(A, Bt, C, bias, M, N, K);
    };

    // 1. activations -> bf16; weights -> bf16 TRANSPOSED [N,K]
    k_cvt_bf16<<<(unsigned)((MH + 255) / 256), 256, 0, stream>>>(x, xb, (int)MH);
    cvtT(Wq,  Wqt, HID, HID);
    cvtT(Wk,  Wkt, HID, HID);
    cvtT(Wv,  Wvt, HID, HID);
    cvtT(Wg,  Wgt, HID, HID);
    cvtT(Wo,  Wot, HID, HID);
    cvtT(gw1, g1t, 2 * HID, HID);
    cvtT(gw2, g2t, HID, HID * WW);

    // 2. projections (WMMA)
    gemm(xb, Wqt, qf,    nullptr, MROWS, HID, HID);
    gemm(xb, Wkt, kf,    nullptr, MROWS, HID, HID);
    gemm(xb, Wvt, vf,    nullptr, MROWS, HID, HID);
    gemm(xb, Wgt, gatef, nullptr, MROWS, HID, HID);

    // 3. kernel generator: gi = concat(q,k); h1 = silu(gi@gen_w1); kern = h1@gen_w2 + b2
    k_concat_qk<<<(unsigned)((MH + 255) / 256), 256, 0, stream>>>(qf, kf, gi);
    gemm(gi, g1t, h1f, nullptr, MROWS, HID, 2 * HID);
    k_silu_cvt_bf16<<<(unsigned)((MH + 255) / 256), 256, 0, stream>>>(h1f, h1b, (int)MH);
    gemm(h1b, g2t, kernf, gb2, MROWS, HID * WW, HID);

    // 4. dynamic causal conv + silu
    k_conv_silu<<<(unsigned)((MH + 255) / 256), 256, 0, stream>>>(kernf, vf, vs);

    // 5. gating params (N=16 -> VALU) and qk L2 norm (in place)
    k_gating<<<(MROWS * HH + 255) / 256, 256, 0, stream>>>(x, Wb, Wa, dtb, Alog, betaf, decf);
    k_l2norm<<<(MROWS * HH + 7) / 8, 256, 0, stream>>>(qf, MROWS * HH);
    k_l2norm<<<(MROWS * HH + 7) / 8, 256, 0, stream>>>(kf, MROWS * HH);

    // 6. sequential gated delta rule, one WG per (b,h), state in LDS
    k_recurrence<<<BB * HH, 128, 0, stream>>>(qf, kf, vs, decf, betaf, of);

    // 7. gated RMSNorm -> bf16, output projection (WMMA)
    k_rmsnorm_gate<<<(MROWS * HH + 7) / 8, 256, 0, stream>>>(of, gatef, nw, yb, MROWS * HH);
    gemm(yb, Wot, out, nullptr, MROWS, HID, HID);
}
